// GAT_21303037788662
// MI455X (gfx1250) — matrix operand order repaired and checked by reading, not compiled, not run on hardware
//
#include <hip/hip_runtime.h>
#include <hip/hip_bf16.h>
#include <math.h>

typedef __attribute__((ext_vector_type(16))) _Float16 v16h;
typedef __attribute__((ext_vector_type(8)))  float    v8f;

#define F_IN 128
#define NHEAD 4
#define DHEAD 64
#define HD    256   // NHEAD * DHEAD
#define NEG_SLOPE 0.2f

// ---------------------------------------------------------------------------
// Kernel 1: h = feats @ W  via WMMA f16->f32.
// Block = 128 threads (4 waves). blockIdx.x = 16-row M tile, blockIdx.y*4+wave
// selects the 16-col N tile.  A tile (16x128 f32 = 8KB) is DMA'd to LDS with
// GLOBAL_LOAD_ASYNC_TO_LDS_B128 (ASYNCcnt path), B gathered from global with
// f32->f16 convert.  4 x v_wmma_f32_16x16x32_f16 per wave (K=128).
// ---------------------------------------------------------------------------
__global__ void gat_gemm_wmma(const float* __restrict__ A,   // [N, 128]
                              const float* __restrict__ W,   // [128, 256]
                              float* __restrict__ Hp,        // [N, 256]
                              int N) {
  __shared__ __align__(16) float Asf[16 * F_IN];   // 8 KB f32 staging tile

  const int tid   = threadIdx.x;
  const int lane  = tid & 31;
  const int row0  = blockIdx.x * 16;
  const int ntile = blockIdx.y * 4 + (tid >> 5);

  if (row0 + 16 <= N) {
    // Async DMA: whole tile is contiguous (row-major [N,128]); each of the
    // 128 threads moves 4 x 16B.  Each wave drains its own ASYNCcnt, then the
    // workgroup barrier orders all LDS writes.
    const uint32_t lds_base = (uint32_t)(uintptr_t)(void*)Asf;
    const uint64_t gbase    = (uint64_t)(uintptr_t)(A + (size_t)row0 * F_IN);
    #pragma unroll
    for (int it = 0; it < 4; ++it) {
      const uint32_t byteoff = (uint32_t)(it * 2048 + tid * 16);
      const uint32_t ldsa = lds_base + byteoff;
      const uint64_t ga   = gbase + byteoff;
      asm volatile("global_load_async_to_lds_b128 %0, %1, off"
                   :: "v"(ldsa), "v"(ga) : "memory");
    }
    asm volatile("s_wait_asynccnt 0x0" ::: "memory");
  } else {
    // Ragged M tail (not hit for N % 16 == 0): guarded scalar staging.
    for (int i = tid; i < 16 * F_IN; i += 128) {
      const int r = i >> 7, c = i & (F_IN - 1);
      Asf[i] = (row0 + r < N) ? A[(size_t)(row0 + r) * F_IN + c] : 0.0f;
    }
  }
  __syncthreads();

  const int half16 = lane >> 4;                 // 0: lanes 0-15, 1: lanes 16-31
  const int mrow   = lane & 15;                 // A row owned by this lane
  const int ncol   = ntile * 16 + (lane & 15);  // B/D column owned by this lane

  v8f acc = {};
  #pragma unroll
  for (int kk = 0; kk < F_IN / 32; ++kk) {
    // A 16x32 f16 layout (ISA 7.12.2): lane<16 holds M=lane, K={0..7,16..23};
    // lanes>=16 hold the same rows with K offset +8.
    v16h a;
    const int kbaseA = kk * 32 + (half16 ? 8 : 0);
    #pragma unroll
    for (int i = 0; i < 16; ++i) {
      const int k = kbaseA + i + ((i >= 8) ? 8 : 0);
      a[i] = (_Float16)Asf[mrow * F_IN + k];
    }
    // B 32x16 f16 layout (same K order as A): lanes 0-15 hold K={0..7,16..23}
    // of column `lane`, lanes 16-31 hold K={8..15,24..31} of column `lane-16`.
    v16h b;
    const int kbaseB = kk * 32 + (half16 ? 8 : 0);
    #pragma unroll
    for (int i = 0; i < 16; ++i) {
      b[i] = (_Float16)W[(kbaseB + i + ((i >= 8) ? 8 : 0)) * HD + ncol];
    }
    acc = __builtin_amdgcn_wmma_f32_16x16x32_f16(
        /*neg_a=*/false, a, /*neg_b=*/false, b,
        /*c_mod=*/(short)0, acc, /*reuse_a=*/false, /*reuse_b=*/false);
  }

  // D 16x16 f32: VGPR r holds M=r (lanes 0-15) / M=r+8 (lanes 16-31).
  // Tile-level guard: no per-row branches in the common (full-tile) case.
  if (row0 + 16 <= N) {
    #pragma unroll
    for (int r = 0; r < 8; ++r)
      Hp[(size_t)(row0 + r + (half16 ? 8 : 0)) * HD + ncol] = acc[r];
  } else {
    #pragma unroll
    for (int r = 0; r < 8; ++r) {
      const int row = row0 + r + (half16 ? 8 : 0);
      if (row < N) Hp[(size_t)row * HD + ncol] = acc[r];
    }
  }
}

// ---------------------------------------------------------------------------
// Kernel 2: el[n,h] = <h[n,h,:], attn_l[h,:]>, er likewise.
// ---------------------------------------------------------------------------
__global__ void gat_attn_dots(const float* __restrict__ Hp,
                              const float* __restrict__ attn_l,
                              const float* __restrict__ attn_r,
                              float* __restrict__ el,
                              float* __restrict__ er, int N) {
  const int idx = blockIdx.x * blockDim.x + threadIdx.x;  // node*H + head
  if (idx >= N * NHEAD) return;
  const int node = idx >> 2;
  const int head = idx & (NHEAD - 1);
  const float* hp = Hp + (size_t)node * HD + head * DHEAD;
  const float* al = attn_l + head * DHEAD;
  const float* ar = attn_r + head * DHEAD;
  float sl = 0.0f, sr = 0.0f;
  #pragma unroll 8
  for (int d = 0; d < DHEAD; ++d) {
    const float v = hp[d];
    sl += v * al[d];
    sr += v * ar[d];
  }
  el[idx] = sl;
  er[idx] = sr;
}

// ---------------------------------------------------------------------------
// Kernel 3: init softmax state each call (m = -inf, z = 0, accum = 0).
// ---------------------------------------------------------------------------
__global__ void gat_init(float* __restrict__ m, float* __restrict__ z,
                         float* __restrict__ accum, int N) {
  const int idx = blockIdx.x * blockDim.x + threadIdx.x;
  if (idx < N * NHEAD) {
    m[idx] = -__builtin_inff();
    z[idx] = 0.0f;
  }
  if (idx < N * HD) accum[idx] = 0.0f;
}

__device__ __forceinline__ float leaky(float x) {
  return x > 0.0f ? x : NEG_SLOPE * x;
}

// Ordered float atomic-max via signed/unsigned integer monotone mapping.
__device__ __forceinline__ void atomicMaxF32(float* addr, float v) {
  if (v >= 0.0f) atomicMax((int*)addr, __float_as_int(v));
  else           atomicMin((unsigned int*)addr, __float_as_uint(v));
}

// ---------------------------------------------------------------------------
// Kernel 4: segment max of edge scores into m[dst,h].
// ---------------------------------------------------------------------------
__global__ void gat_edge_max(const int* __restrict__ src, const int* __restrict__ dst,
                             const float* __restrict__ el, const float* __restrict__ er,
                             float* __restrict__ m, int E) {
  const int idx = blockIdx.x * blockDim.x + threadIdx.x;  // edge*H + head
  if (idx >= E * NHEAD) return;
  const int e = idx >> 2;
  const int h = idx & (NHEAD - 1);
  const int s = src[e], d = dst[e];
  const float v = leaky(el[s * NHEAD + h] + er[d * NHEAD + h]);
  atomicMaxF32(&m[d * NHEAD + h], v);
}

// ---------------------------------------------------------------------------
// Kernel 5: z[dst,h] += exp(e - m[dst,h])  (scores recomputed, L2-resident).
// ---------------------------------------------------------------------------
__global__ void gat_edge_expsum(const int* __restrict__ src, const int* __restrict__ dst,
                                const float* __restrict__ el, const float* __restrict__ er,
                                const float* __restrict__ m, float* __restrict__ z, int E) {
  const int idx = blockIdx.x * blockDim.x + threadIdx.x;
  if (idx >= E * NHEAD) return;
  const int e = idx >> 2;
  const int h = idx & (NHEAD - 1);
  const int s = src[e], d = dst[e];
  const float v  = leaky(el[s * NHEAD + h] + er[d * NHEAD + h]);
  const float ex = __expf(v - m[d * NHEAD + h]);
  atomicAdd(&z[d * NHEAD + h], ex);
}

// ---------------------------------------------------------------------------
// Kernel 6: accum[dst,h,d] += alpha * h[src,h,d].  One block per edge,
// 256 threads cover all (h,d).  src/dst are block-uniform -> scalar loads;
// the h[src] row (the roofline-dominant ~800 MB of gather traffic) is
// prefetched into the WGP cache while alpha is computed.
// ---------------------------------------------------------------------------
__global__ void gat_edge_aggregate(const int* __restrict__ src, const int* __restrict__ dst,
                                   const float* __restrict__ el, const float* __restrict__ er,
                                   const float* __restrict__ m, const float* __restrict__ z,
                                   const float* __restrict__ Hp, float* __restrict__ accum,
                                   int E) {
  const int e = blockIdx.x;               // grid.x == E exactly
  __shared__ float alpha_s[NHEAD];
  const int s = src[e];                   // uniform -> s_load
  const int d = dst[e];
  // Warm the cache for the gather (global_prefetch_b8) while alpha is built.
  __builtin_prefetch(&Hp[(size_t)s * HD + threadIdx.x], 0, 3);
  if (threadIdx.x < NHEAD) {
    const int h = threadIdx.x;
    const float v = leaky(el[s * NHEAD + h] + er[d * NHEAD + h]);
    alpha_s[h] = __expf(v - m[d * NHEAD + h]) / z[d * NHEAD + h];
  }
  __syncthreads();
  const int hd = threadIdx.x;             // 0..255
  const int h  = hd >> 6;                 // /DHEAD
  atomicAdd(&accum[(size_t)d * HD + hd], alpha_s[h] * Hp[(size_t)s * HD + hd]);
}

// ---------------------------------------------------------------------------
// Kernel 7: out[n,d] = relu(mean_h(accum[n,h,d] + bias[h,d])).
// ---------------------------------------------------------------------------
__global__ void gat_finalize(const float* __restrict__ accum,
                             const float* __restrict__ bias,
                             float* __restrict__ out, int N) {
  const int idx = blockIdx.x * blockDim.x + threadIdx.x;  // node*D + d
  if (idx >= N * DHEAD) return;
  const int n = idx >> 6;
  const int d = idx & (DHEAD - 1);
  float sacc = 0.0f;
  #pragma unroll
  for (int h = 0; h < NHEAD; ++h)
    sacc += accum[(size_t)n * HD + h * DHEAD + d] + bias[h * DHEAD + d];
  const float r = sacc * (1.0f / NHEAD);
  out[idx] = r > 0.0f ? r : 0.0f;
}

extern "C" void kernel_launch(void* const* d_in, const int* in_sizes, int n_in,
                              void* d_out, int out_size, void* d_ws, size_t ws_size,
                              hipStream_t stream) {
  const float* feats  = (const float*)d_in[0];
  const int*   src    = (const int*)  d_in[1];
  const int*   dst    = (const int*)  d_in[2];
  const float* W      = (const float*)d_in[3];
  const float* attn_l = (const float*)d_in[4];
  const float* attn_r = (const float*)d_in[5];
  const float* bias   = (const float*)d_in[6];
  float* out = (float*)d_out;

  const int N = in_sizes[0] / F_IN;   // 50000
  const int E = in_sizes[1];          // 800000

  // Workspace layout (floats): ~106 MB total.
  float* ws    = (float*)d_ws;
  float* Hp    = ws;                       // N*256
  float* el    = Hp   + (size_t)N * HD;    // N*4
  float* er    = el   + (size_t)N * NHEAD; // N*4
  float* m     = er   + (size_t)N * NHEAD; // N*4
  float* z     = m    + (size_t)N * NHEAD; // N*4
  float* accum = z    + (size_t)N * NHEAD; // N*256

  // 1) WMMA GEMM
  {
    dim3 grid((N + 15) / 16, HD / (4 * 16));
    gat_gemm_wmma<<<grid, 128, 0, stream>>>(feats, W, Hp, N);
  }
  // 2) attention dots
  {
    const int total = N * NHEAD;
    gat_attn_dots<<<(total + 255) / 256, 256, 0, stream>>>(Hp, attn_l, attn_r, el, er, N);
  }
  // 3) init m / z / accum
  {
    const int total = N * HD;
    gat_init<<<(total + 255) / 256, 256, 0, stream>>>(m, z, accum, N);
  }
  // 4) segment max
  {
    const int total = E * NHEAD;
    gat_edge_max<<<(total + 255) / 256, 256, 0, stream>>>(src, dst, el, er, m, E);
  }
  // 5) segment exp-sum
  {
    const int total = E * NHEAD;
    gat_edge_expsum<<<(total + 255) / 256, 256, 0, stream>>>(src, dst, el, er, m, z, E);
  }
  // 6) weighted scatter-aggregate
  gat_edge_aggregate<<<E, HD, 0, stream>>>(src, dst, el, er, m, z, Hp, accum, E);
  // 7) head-mean + bias + relu
  {
    const int total = N * DHEAD;
    gat_finalize<<<(total + 255) / 256, 256, 0, stream>>>(accum, bias, out, N);
  }
}
